// Actor_80917183857065
// MI455X (gfx1250) — matrix operand order repaired and checked by reading, not compiled, not run on hardware
//
#include <hip/hip_runtime.h>

typedef __attribute__((ext_vector_type(16))) __bf16 v16bf;
typedef __attribute__((ext_vector_type(8)))  float  v8f;

#define B_   4
#define N_   2000
#define E_   32000
#define FIN_ 64
#define CE_  64
#define CH_  128

// ---------------------------------------------------------------------------
// float -> bf16 round-to-nearest-even (bit twiddle; no scalar __bf16 conv).
// ---------------------------------------------------------------------------
__device__ __forceinline__ unsigned short f2bf(float f) {
  unsigned u = __float_as_uint(f);
  u += 0x7FFFu + ((u >> 16) & 1u);
  return (unsigned short)(u >> 16);
}

// ---------------------------------------------------------------------------
// Generic fused GEMM:
//   C[M x Nout] = post( concatA[M x K] @ W[K x Nout] + bias )
// concatA row m = [ A0[idx0?idx0[m]:m] | A1[...] | A2[...] ]  (row gather)
// post: optional ReLU, optional per-row scale, optional scatter-add via
// out_idx (fuses jax.ops.segment_sum into the producing GEMM).
//
// bf16 WMMA 16x16x32, f32 accumulate. Block tile 128x64 (M x N), KT=32,
// 4 waves; each wave owns a 32x64 sub-tile: 2 A fragments, 4 B fragments
// reused across 2 WMMAs each -> 8 WMMAs per K-step, 6 DS dwords per WMMA.
// All K here are multiples of 32; segment widths are multiples of 64, so a
// 32-aligned 32-wide column window never crosses a segment boundary.
// ---------------------------------------------------------------------------
__global__ __launch_bounds__(128) void gemm_wmma_kernel(
    const float* __restrict__ A0, const int* __restrict__ idx0, int k0,
    const float* __restrict__ A1, const int* __restrict__ idx1, int k1,
    const float* __restrict__ A2, const int* __restrict__ idx2, int k2,
    const float* __restrict__ W,  const float* __restrict__ bias,
    float* __restrict__ C, const int* __restrict__ out_idx,
    const float* __restrict__ row_scale,
    int M, int Nout, int relu)
{
  __shared__ __attribute__((aligned(16))) unsigned short As[128][34];
  __shared__ __attribute__((aligned(16))) unsigned short Bs[64][34]; // [n][k]

  const int tid  = threadIdx.x;
  const int wave = tid >> 5;
  const int lane = tid & 31;
  const int r16  = lane & 15;
  const int hi   = lane >> 4;
  const int m0   = blockIdx.x * 128;
  const int n0   = blockIdx.y * 64;
  const int K    = k0 + k1 + k2;

  v8f acc[8] = {};   // [g*4+j]: g = row-half (0/1), j = 16-col group

  const int bk  = tid >> 2;          // B tile: k 0..31
  const int bnb = (tid & 3) << 4;    //         n base 0,16,32,48

  for (int kt = 0; kt < K; kt += 32) {
    // ---- stage A tile (128 x 32) as bf16; one row per thread ----
    {
      const int m = m0 + tid;
      const float* src; int off, ks; const int* ix;
      if (kt < k0)           { src = A0; off = 0;       ks = k0; ix = idx0; }
      else if (kt < k0 + k1) { src = A1; off = k0;      ks = k1; ix = idx1; }
      else                   { src = A2; off = k0 + k1; ks = k2; ix = idx2; }
      if (m < M) {
        const int rr = ix ? ix[m] : m;
        const float* p = src + (size_t)rr * ks + (kt - off);
#pragma unroll
        for (int t = 0; t < 32; ++t) As[tid][t] = f2bf(p[t]);
      } else {
#pragma unroll
        for (int t = 0; t < 32; ++t) As[tid][t] = 0;
      }
    }
    // ---- stage B tile (32 x 64) transposed to Bs[n][k] ----
    {
      const int kg = kt + bk;
#pragma unroll
      for (int t = 0; t < 16; ++t) {
        const int n = bnb + t;
        float v = (n0 + n < Nout) ? W[(size_t)kg * Nout + (n0 + n)] : 0.f;
        Bs[n][bk] = f2bf(v);
      }
    }
    __syncthreads();

    // ---- fragments per ISA lane layouts, then 8 WMMAs ----
    union UF { v16bf v; unsigned u[8]; };
    UF a0, a1;
#pragma unroll
    for (int vv = 0; vv < 8; ++vv) {
      // A 16x32 bf16: lanes 0-15 hold K 0..7,16..23; lanes 16-31 K 8..15,24..31
      const int kk = 2 * vv + 8 * hi + ((vv >= 4) ? 8 : 0);
      a0.u[vv] = *(const unsigned*)&As[wave * 32 + r16][kk];
      a1.u[vv] = *(const unsigned*)&As[wave * 32 + 16 + r16][kk];
    }
#pragma unroll
    for (int j = 0; j < 4; ++j) {
      UF bf;
#pragma unroll
      for (int vv = 0; vv < 8; ++vv) {
        // B 32x16 bf16: lanes 0-15 hold K 0..7,16..23; lanes 16-31 K 8..15,24..31
        const int kk = 2 * vv + 8 * hi + ((vv >= 4) ? 8 : 0);
        bf.u[vv] = *(const unsigned*)&Bs[j * 16 + r16][kk];
      }
      acc[j]     = __builtin_amdgcn_wmma_f32_16x16x32_bf16(
          false, a0.v, false, bf.v, (short)0, acc[j], false, false);
      acc[4 + j] = __builtin_amdgcn_wmma_f32_16x16x32_bf16(
          false, a1.v, false, bf.v, (short)0, acc[4 + j], false, false);
    }
    __syncthreads();
  }

  // ---- epilogue: bias + relu + row scale + (store | scatter-add) ----
#pragma unroll
  for (int j = 0; j < 4; ++j) {
    const int n = n0 + j * 16 + r16;
    if (n >= Nout) continue;
    const float bv = bias ? bias[n] : 0.f;
#pragma unroll
    for (int g = 0; g < 2; ++g) {
#pragma unroll
      for (int vv = 0; vv < 8; ++vv) {
        const int m = m0 + wave * 32 + g * 16 + vv + 8 * hi;
        if (m >= M) continue;
        float val = acc[g * 4 + j][vv] + bv;
        if (relu) val = fmaxf(val, 0.f);
        if (row_scale) val *= row_scale[m];
        if (out_idx) {
          float* dst = &C[(size_t)out_idx[m] * Nout + n];
          __hip_atomic_fetch_add(dst, val, __ATOMIC_RELAXED,
                                 __HIP_MEMORY_SCOPE_AGENT);
        } else {
          C[(size_t)m * Nout + n] = val;
        }
      }
    }
  }
}

// ---------------------------------------------------------------------------
__global__ void fill_zero_kernel(float* p, int n) {
  int i = blockIdx.x * blockDim.x + threadIdx.x;
  if (i < n) p[i] = 0.f;
}

// ii[0..E) = a0; ii[E..2E) = a1 + N   (fold rec/lig intra graphs into one GEMM)
__global__ void build_idx_kernel(const int* __restrict__ a0,
                                 const int* __restrict__ a1,
                                 int* __restrict__ out, int E, int N) {
  int e = blockIdx.x * blockDim.x + threadIdx.x;
  if (e < E) { out[e] = a0[e]; out[E + e] = a1[e] + N; }
}

// cat[2N x 256]: rows 0..N-1 = [F_r | p2], rows N..2N-1 = [F_l | Pp1]
__global__ void build_cat_kernel(const float* __restrict__ F,
                                 const float* __restrict__ p2,
                                 const float* __restrict__ Pp1,
                                 float* __restrict__ cat, int N) {
  int i = blockIdx.x * blockDim.x + threadIdx.x;
  if (i >= 2 * N * 256) return;
  int r = i >> 8, c = i & 255;
  float v;
  if (c < 128)      v = F[(size_t)r * 128 + c];
  else if (r < N)   v = p2[(size_t)r * 128 + (c - 128)];
  else              v = Pp1[(size_t)(r - N) * 128 + (c - 128)];
  cat[i] = v;
}

// out = softmax(logits * mask) over n elements (single block)
__global__ __launch_bounds__(256) void softmax_mask_kernel(
    const float* __restrict__ lg, const float* __restrict__ mask,
    float* __restrict__ out, int n) {
  __shared__ float red[256];
  const int tid = threadIdx.x;
  float mx = -3.402823466e38f;
  for (int i = tid; i < n; i += 256) mx = fmaxf(mx, lg[i] * mask[i]);
  red[tid] = mx; __syncthreads();
  for (int s = 128; s > 0; s >>= 1) {
    if (tid < s) red[tid] = fmaxf(red[tid], red[tid + s]);
    __syncthreads();
  }
  const float gmax = red[0];
  __syncthreads();
  float sum = 0.f;
  for (int i = tid; i < n; i += 256) sum += expf(lg[i] * mask[i] - gmax);
  red[tid] = sum; __syncthreads();
  for (int s = 128; s > 0; s >>= 1) {
    if (tid < s) red[tid] += red[tid + s];
    __syncthreads();
  }
  const float inv = 1.f / red[0];
  for (int i = tid; i < n; i += 256) out[i] = expf(lg[i] * mask[i] - gmax) * inv;
}

// ---------------------------------------------------------------------------
struct Dense { const float* w; const float* b; };
struct Mlp2  { Dense l0, l1; };

static inline void launch_gemm(hipStream_t st,
    const float* A0, const int* x0, int k0,
    const float* A1, const int* x1, int k1,
    const float* A2, const int* x2, int k2,
    Dense d, int M, int Nout, float* C, int relu,
    const int* out_idx, const float* row_scale) {
  dim3 grid((M + 127) / 128, (Nout + 63) / 64);
  gemm_wmma_kernel<<<grid, dim3(128), 0, st>>>(
      A0, x0, k0, A1, x1, k1, A2, x2, k2,
      d.w, d.b, C, out_idx, row_scale, M, Nout, relu);
}

static inline void zero_f(hipStream_t st, float* p, int n) {
  fill_zero_kernel<<<(n + 255) / 256, 256, 0, st>>>(p, n);
}

extern "C" void kernel_launch(void* const* d_in, const int* in_sizes, int n_in,
                              void* d_out, int out_size, void* d_ws, size_t ws_size,
                              hipStream_t stream) {
  (void)n_in; (void)out_size;

  // ---- top-level input order: insertion (masks first) vs sorted (edges first)
  const bool ins = (in_sizes[0] == B_ * 6 * N_);
  const float *masks, *nodes, *edges;
  const int *i_s, *j_s;
  if (ins) {
    masks = (const float*)d_in[0]; nodes = (const float*)d_in[1];
    edges = (const float*)d_in[2];
    i_s = (const int*)d_in[3]; j_s = (const int*)d_in[4];
  } else {
    edges = (const float*)d_in[0];
    i_s = (const int*)d_in[1]; j_s = (const int*)d_in[2];
    masks = (const float*)d_in[3]; nodes = (const float*)d_in[4];
  }

  // ---- parameter leaves (both flatten orders supported) ----
  int li = 5;
  auto nf = [&]() { return (const float*)d_in[li++]; };
  Mlp2 node_input, Pp1_merge, P_join, p1_join, p2_join, P_out, p1_out, p2_out;
  Mlp2 intra_msg[6], intra_node[6], inter_msg[6], inter_node[6], inter_edge[6];
  if (ins) {  // insertion order: dict insertion, dense = {w, b}
    auto D  = [&](Dense& d) { d.w = nf(); d.b = nf(); };
    auto Ml = [&](Mlp2& m)  { D(m.l0); D(m.l1); };
    Ml(node_input); Ml(Pp1_merge);
    Ml(P_join); Ml(p1_join); Ml(p2_join);
    Ml(P_out);  Ml(p1_out);  Ml(p2_out);
    for (int m = 0; m < 6; ++m) { Ml(intra_msg[m]); Ml(intra_node[m]); }
    for (int m = 0; m < 6; ++m) { Ml(inter_msg[m]); Ml(inter_node[m]); Ml(inter_edge[m]); }
  } else {    // jax-sorted keys: dense = {b, w}
    auto D  = [&](Dense& d) { d.b = nf(); d.w = nf(); };
    auto Ml = [&](Mlp2& m)  { D(m.l0); D(m.l1); };
    Ml(P_join); Ml(P_out); Ml(Pp1_merge);
    for (int m = 0; m < 6; ++m) { Ml(inter_edge[m]); Ml(inter_msg[m]); Ml(inter_node[m]); }
    for (int m = 0; m < 6; ++m) { Ml(intra_msg[m]); Ml(intra_node[m]); }
    Ml(node_input); Ml(p1_join); Ml(p1_out); Ml(p2_join); Ml(p2_out);
  }

  // ---- workspace carve (~54 MB, reused across the 4 batch elements) ----
  char* wp = (char*)d_ws;
  auto carve = [&](size_t bytes) {
    char* p = wp; wp += (bytes + 255) & ~(size_t)255; return p;
  };
  float* F    = (float*)carve((size_t)2 * N_ * CH_ * 4);      // node feats [2N,128]
  float* cat  = (float*)carve((size_t)2 * N_ * 2 * CH_ * 4);  // [2N,256]
  float* Pb   = (float*)carve((size_t)N_ * CH_ * 4);
  float* p1b  = (float*)carve((size_t)N_ * CH_ * 4);
  float* p2b  = (float*)carve((size_t)N_ * CH_ * 4);
  float* Pp1b = (float*)carve((size_t)N_ * CH_ * 4);
  float* hid  = (float*)carve((size_t)2 * E_ * CH_ * 4);      // MLP hidden [2E,128]
  float* agg  = (float*)carve((size_t)2 * N_ * CH_ * 4);
  float* eib  = (float*)carve((size_t)E_ * CE_ * 4);          // inter edge feats
  int*   ii   = (int*)  carve((size_t)2 * E_ * 4);
  int*   jj   = (int*)  carve((size_t)2 * E_ * 4);
  float* lgt  = (float*)carve((size_t)N_ * 4);
  if ((size_t)(wp - (char*)d_ws) > ws_size) return;

  const int*   NI  = nullptr;
  const float* NFp = nullptr;
  const float* nlF = F + (size_t)N_ * CH_;  // ligand rows

  for (int b = 0; b < B_; ++b) {
    const float* mask_b  = masks + (size_t)b * 6 * N_;
    const float* int_l   = mask_b + 3 * N_;
    const float* rim_r   = mask_b + 4 * N_;
    const float* rim_l   = mask_b + 5 * N_;
    const float* nodes_b = nodes + (size_t)b * 2 * N_ * FIN_;
    const float* edges_b = edges + (size_t)b * 3 * E_ * CE_;  // edge0|edge1|edge2
    const int* i0 = i_s + (size_t)b * 3 * E_;
    const int* j0 = j_s + (size_t)b * 3 * E_;
    const int* i2 = i0 + 2 * E_;
    const int* j2 = j0 + 2 * E_;

    build_idx_kernel<<<(E_ + 255) / 256, 256, 0, stream>>>(i0, i0 + E_, ii, E_, N_);
    build_idx_kernel<<<(E_ + 255) / 256, 256, 0, stream>>>(j0, j0 + E_, jj, E_, N_);
    hipMemcpyAsync(eib, edges_b + (size_t)2 * E_ * CE_, (size_t)E_ * CE_ * 4,
                   hipMemcpyDeviceToDevice, stream);
    zero_f(stream, Pb, N_ * CH_);
    zero_f(stream, p1b, N_ * CH_);
    zero_f(stream, p2b, N_ * CH_);

    // node_input MLP on both rec+lig rows at once (contiguous [2N,64])
    launch_gemm(stream, nodes_b, NI, FIN_, NFp, NI, 0, NFp, NI, 0,
                node_input.l0, 2 * N_, 64, hid, 1, NI, NFp);
    launch_gemm(stream, hid, NI, 64, NFp, NI, 0, NFp, NI, 0,
                node_input.l1, 2 * N_, CH_, F, 0, NI, NFp);

    for (int n = 2; n <= 6; n += 2) {
      // Pp1 = Pp1_merge([P | p1])
      launch_gemm(stream, Pb, NI, CH_, p1b, NI, CH_, NFp, NI, 0,
                  Pp1_merge.l0, N_, CH_, hid, 1, NI, NFp);
      launch_gemm(stream, hid, NI, CH_, NFp, NI, 0, NFp, NI, 0,
                  Pp1_merge.l1, N_, CH_, Pp1b, 0, NI, NFp);
      // cat = [F | p2/Pp1]  (din 128 -> 256)
      build_cat_kernel<<<(2 * N_ * 256 + 255) / 256, 256, 0, stream>>>(
          F, p2b, Pp1b, cat, N_);

      // ---- intra layer m = n-2 (din = 256), rec+lig fused (M = 2E) ----
      {
        Dense ml0 = intra_msg[n - 2].l0, ml1 = intra_msg[n - 2].l1;
        Dense nl0 = intra_node[n - 2].l0, nl1 = intra_node[n - 2].l1;
        launch_gemm(stream, cat, ii, 2 * CH_, cat, jj, 2 * CH_, edges_b, NI, CE_,
                    ml0, 2 * E_, CH_, hid, 1, NI, NFp);
        zero_f(stream, agg, 2 * N_ * CH_);
        launch_gemm(stream, hid, NI, CH_, NFp, NI, 0, NFp, NI, 0,
                    ml1, 2 * E_, CH_, agg, 0, ii, NFp);        // fused segment_sum
        launch_gemm(stream, cat, NI, 2 * CH_, agg, NI, CH_, NFp, NI, 0,
                    nl0, 2 * N_, CH_, hid, 1, NI, NFp);
        launch_gemm(stream, hid, NI, CH_, NFp, NI, 0, NFp, NI, 0,
                    nl1, 2 * N_, CH_, F, 0, NI, NFp);
      }
      // ---- intra layer m = n-1 (din = 128) ----
      {
        Dense ml0 = intra_msg[n - 1].l0, ml1 = intra_msg[n - 1].l1;
        Dense nl0 = intra_node[n - 1].l0, nl1 = intra_node[n - 1].l1;
        launch_gemm(stream, F, ii, CH_, F, jj, CH_, edges_b, NI, CE_,
                    ml0, 2 * E_, CH_, hid, 1, NI, NFp);
        zero_f(stream, agg, 2 * N_ * CH_);
        launch_gemm(stream, hid, NI, CH_, NFp, NI, 0, NFp, NI, 0,
                    ml1, 2 * E_, CH_, agg, 0, ii, NFp);
        launch_gemm(stream, F, NI, CH_, agg, NI, CH_, NFp, NI, 0,
                    nl0, 2 * N_, CH_, hid, 1, NI, NFp);
        launch_gemm(stream, hid, NI, CH_, NFp, NI, 0, NFp, NI, 0,
                    nl1, 2 * N_, CH_, F, 0, NI, NFp);
      }
      // ---- inter layers m = n-2, n-1 (node+edge update, M = E, 2N nodes) ----
      for (int m = n - 2; m < n; ++m) {
        Dense ml0 = inter_msg[m].l0, ml1 = inter_msg[m].l1;
        Dense el0 = inter_edge[m].l0, el1 = inter_edge[m].l1;
        Dense nl0 = inter_node[m].l0, nl1 = inter_node[m].l1;
        launch_gemm(stream, F, i2, CH_, F, j2, CH_, eib, NI, CE_,
                    ml0, E_, CH_, hid, 1, NI, NFp);
        zero_f(stream, agg, 2 * N_ * CH_);
        launch_gemm(stream, hid, NI, CH_, NFp, NI, 0, NFp, NI, 0,
                    ml1, E_, CH_, agg, 0, i2, NFp);
        launch_gemm(stream, F, i2, CH_, F, j2, CH_, eib, NI, CE_,
                    el0, E_, CH_, hid, 1, NI, NFp);
        launch_gemm(stream, hid, NI, CH_, NFp, NI, 0, NFp, NI, 0,
                    el1, E_, CE_, eib, 0, NI, NFp);            // in-place ei
        launch_gemm(stream, F, NI, CH_, agg, NI, CH_, NFp, NI, 0,
                    nl0, 2 * N_, CH_, hid, 1, NI, NFp);
        launch_gemm(stream, hid, NI, CH_, NFp, NI, 0, NFp, NI, 0,
                    nl1, 2 * N_, CH_, F, 0, NI, NFp);          // in-place F
      }
      // ---- joins (masked) ----
      launch_gemm(stream, nlF, NI, CH_, NFp, NI, 0, NFp, NI, 0,
                  P_join.l0, N_, CH_, hid, 1, NI, NFp);
      launch_gemm(stream, hid, NI, CH_, NFp, NI, 0, NFp, NI, 0,
                  P_join.l1, N_, CH_, Pb, 0, NI, int_l);
      launch_gemm(stream, nlF, NI, CH_, NFp, NI, 0, NFp, NI, 0,
                  p1_join.l0, N_, CH_, hid, 1, NI, NFp);
      launch_gemm(stream, hid, NI, CH_, NFp, NI, 0, NFp, NI, 0,
                  p1_join.l1, N_, CH_, p1b, 0, NI, rim_l);
      launch_gemm(stream, F, NI, CH_, NFp, NI, 0, NFp, NI, 0,
                  p2_join.l0, N_, CH_, hid, 1, NI, NFp);
      launch_gemm(stream, hid, NI, CH_, NFp, NI, 0, NFp, NI, 0,
                  p2_join.l1, N_, CH_, p2b, 0, NI, rim_r);
    }

    // ---- output heads + masked softmax ----
    float* outb = (float*)d_out + (size_t)b * 3 * N_;
    launch_gemm(stream, nlF, NI, CH_, NFp, NI, 0, NFp, NI, 0,
                P_out.l0, N_, 64, hid, 1, NI, NFp);
    launch_gemm(stream, hid, NI, 64, NFp, NI, 0, NFp, NI, 0,
                P_out.l1, N_, 1, lgt, 0, NI, NFp);
    softmax_mask_kernel<<<1, 256, 0, stream>>>(lgt, int_l, outb, N_);
    launch_gemm(stream, nlF, NI, CH_, NFp, NI, 0, NFp, NI, 0,
                p1_out.l0, N_, 64, hid, 1, NI, NFp);
    launch_gemm(stream, hid, NI, 64, NFp, NI, 0, NFp, NI, 0,
                p1_out.l1, N_, 1, lgt, 0, NI, NFp);
    softmax_mask_kernel<<<1, 256, 0, stream>>>(lgt, rim_l, outb + N_, N_);
    launch_gemm(stream, F, NI, CH_, NFp, NI, 0, NFp, NI, 0,
                p2_out.l0, N_, 64, hid, 1, NI, NFp);
    launch_gemm(stream, hid, NI, 64, NFp, NI, 0, NFp, NI, 0,
                p2_out.l1, N_, 1, lgt, 0, NI, NFp);
    softmax_mask_kernel<<<1, 256, 0, stream>>>(lgt, rim_r, outb + 2 * N_, N_);
  }
}
